// NestedFeedForward_41644002902117
// MI455X (gfx1250) — matrix operand order repaired and checked by reading, not compiled, not run on hardware
//
#include <hip/hip_runtime.h>
#include <hip/hip_bf16.h>
#include <math.h>

// ---------------------------------------------------------------------------
// Nested MoE FFN for MI455X (gfx1250, wave32).
//   h   = gelu((x*mask_in @ W1^T + b1) * mask_hid)     [expand kernel]
//   out = h @ W2^T + b2                                 [contract kernel]
// bf16 WMMA (v_wmma_f32_16x16x32_bf16), fp32 accumulation. Hidden activations
// round-trip through d_ws as bf16 (N*H*2 = 128 MB). Kernel 2 stages the bf16
// h-tile with the Tensor Data Mover (tensor_load_to_lds + s_wait_tensorcnt).
// ---------------------------------------------------------------------------

typedef __bf16 bf16_t;
typedef __attribute__((ext_vector_type(16))) __bf16        v16bf;
typedef __attribute__((ext_vector_type(8)))  float         v8f;
typedef __attribute__((ext_vector_type(4)))  unsigned int  u32x4;
typedef __attribute__((ext_vector_type(8)))  int           i32x8;
typedef __attribute__((ext_vector_type(4)))  int           i32x4;

#define D_DIM   1024
#define H_DIM   4096
#define TM      128          // tokens per block tile
#define TN      128          // output features per block tile
#define TK      64           // staged K per barrier (2 WMMA substeps of 32)
#define LDK     (TK + 8)     // padded LDS row stride in bf16 (144 B, 16B-aligned)
#define THREADS 256          // 8 wave32s

__device__ __forceinline__ float gelu_exact(float t) {
    return 0.5f * t * (1.0f + erff(t * 0.70710678118654752440f));
}

// One 128x128x64 step from staged LDS tiles (two 16x16x32 K-substeps).
// Wave grid: wm in [0,4) covers 32 M-rows, wn in [0,2) covers 64 N-cols.
// A frag (16x32 bf16, ISA 7.12.2): lane<16 -> M=lane, K=[0..7]+[16..23];
// lane>=16 -> M=lane-16, K=[8..15]+[24..31].
// B frag (32x16 bf16): lane<16 -> N=lane, K=[0..7]+[16..23]; lane>=16 -> K=[8..15]+[24..31].
__device__ __forceinline__ void wmma_tile_step(
    const bf16_t (*__restrict__ As)[LDK], const bf16_t (*__restrict__ Bs)[LDK],
    int wm, int wn, int half, int l16, v8f (&acc)[2][4])
{
#pragma unroll
    for (int ks = 0; ks < TK; ks += 32) {
        v16bf afrag[2];
        v16bf bfrag[4];
#pragma unroll
        for (int mi = 0; mi < 2; ++mi) {
            const bf16_t* ap = &As[wm * 32 + mi * 16 + l16][ks];
#pragma unroll
            for (int i = 0; i < 8; ++i) {
                afrag[mi][i]     = ap[half * 8 + i];
                afrag[mi][8 + i] = ap[16 + half * 8 + i];
            }
        }
#pragma unroll
        for (int ni = 0; ni < 4; ++ni) {
            const bf16_t* bp = &Bs[wn * 64 + ni * 16 + l16][ks];
#pragma unroll
            for (int i = 0; i < 8; ++i) {
                bfrag[ni][i]     = bp[half * 8 + i];
                bfrag[ni][8 + i] = bp[16 + half * 8 + i];
            }
        }
#pragma unroll
        for (int mi = 0; mi < 2; ++mi)
#pragma unroll
            for (int ni = 0; ni < 4; ++ni)
                acc[mi][ni] = __builtin_amdgcn_wmma_f32_16x16x32_bf16(
                    false, afrag[mi], false, bfrag[ni],
                    (short)0, acc[mi][ni], false, false);
    }
}

// ---------------------------------------------------------------------------
// Kernel 1: expand.  h[tok, hid] = gelu((x*mask_in) @ W1^T + b1) * mask_hid
// ---------------------------------------------------------------------------
__global__ __launch_bounds__(THREADS)
void nested_ffn_expand(const float* __restrict__ x,
                       const int*   __restrict__ token_mask,
                       const float* __restrict__ W1,
                       const float* __restrict__ b1,
                       bf16_t*      __restrict__ h_out)
{
    __shared__ __align__(16) bf16_t As[TM][LDK];
    __shared__ __align__(16) bf16_t Bs[TN][LDK];
    __shared__ int   s_shift[TM];
    __shared__ float s_bias[TN];
    __shared__ int   s_minshift;

    const int tid      = threadIdx.x;
    const int tok_base = blockIdx.y * TM;
    const int hid_base = blockIdx.x * TN;

    if (tid == 0) s_minshift = 3;
    __syncthreads();
    if (tid < TM) {
        int sh = 3 - token_mask[tok_base + tid];   // expert m -> shift 3-m
        s_shift[tid] = sh;
        atomicMin(&s_minshift, sh);
    }
    if (tid < TN) s_bias[tid] = b1[hid_base + tid];
    __syncthreads();

    const int  minshift = s_minshift;
    const int  kmax     = D_DIM >> minshift;              // max d_in in tile
    const bool live     = hid_base < (H_DIM >> minshift); // any token uses tile?

    const int lane = tid & 31;
    const int half = lane >> 4;
    const int l16  = lane & 15;
    const int wave = tid >> 5;
    const int wm   = wave & 3;
    const int wn   = wave >> 2;

    v8f acc[2][4] = {};

    if (live) {
        // Hoist per-thread staging coordinates and per-row d_in (K-invariant).
        int din[8];
#pragma unroll
        for (int i = 0; i < 8; ++i)
            din[i] = D_DIM >> s_shift[(tid + i * THREADS) >> 4];

        for (int k0 = 0; k0 < kmax; k0 += TK) {
            // Stage A (x, per-token prefix mask) and B (W1), fp32 -> bf16.
#pragma unroll
            for (int i = 0; i < 8; ++i) {
                const int idx = tid + i * THREADS;   // 2048 float4 per tile
                const int row = idx >> 4;
                const int c4  = (idx & 15) << 2;
                const float4 va =
                    *(const float4*)(x + (size_t)(tok_base + row) * D_DIM + k0 + c4);
                const int kg = k0 + c4;
                As[row][c4 + 0] = (bf16_t)(kg + 0 < din[i] ? va.x : 0.f);
                As[row][c4 + 1] = (bf16_t)(kg + 1 < din[i] ? va.y : 0.f);
                As[row][c4 + 2] = (bf16_t)(kg + 2 < din[i] ? va.z : 0.f);
                As[row][c4 + 3] = (bf16_t)(kg + 3 < din[i] ? va.w : 0.f);
                const float4 vb =
                    *(const float4*)(W1 + (size_t)(hid_base + row) * D_DIM + k0 + c4);
                Bs[row][c4 + 0] = (bf16_t)vb.x;
                Bs[row][c4 + 1] = (bf16_t)vb.y;
                Bs[row][c4 + 2] = (bf16_t)vb.z;
                Bs[row][c4 + 3] = (bf16_t)vb.w;
            }
            if (k0 + TK < kmax) {  // global_prefetch_b8 of next K tile
                const int prow = tid >> 1, pcol = (tid & 1) << 5;
                __builtin_prefetch(x  + (size_t)(tok_base + prow) * D_DIM + k0 + TK + pcol, 0, 1);
                __builtin_prefetch(W1 + (size_t)(hid_base + prow) * D_DIM + k0 + TK + pcol, 0, 1);
            }
            __syncthreads();
            wmma_tile_step(As, Bs, wm, wn, half, l16, acc);
            __syncthreads();
        }
    }

    // Epilogue: bias + exact GELU inside the nested hidden slice, 0 outside.
#pragma unroll
    for (int mi = 0; mi < 2; ++mi) {
        const int lm = wm * 32 + mi * 16 + half * 8;
#pragma unroll
        for (int r = 0; r < 8; ++r) {
            const int row  = lm + r;
            const int dhid = H_DIM >> s_shift[row];
            bf16_t* orow   = h_out + (size_t)(tok_base + row) * H_DIM + hid_base;
#pragma unroll
            for (int ni = 0; ni < 4; ++ni) {
                const int ln = wn * 64 + ni * 16 + l16;
                float v = 0.f;
                if (hid_base + ln < dhid)
                    v = gelu_exact(acc[mi][ni][r] + s_bias[ln]);
                orow[ln] = (bf16_t)v;
            }
        }
    }
}

// ---------------------------------------------------------------------------
// Kernel 2: contract.  out[tok, d] = h @ W2^T + b2   (h already nested-zeroed)
// A tile (raw bf16 2D copy) is staged by the Tensor Data Mover.
// ---------------------------------------------------------------------------
__global__ __launch_bounds__(THREADS)
void nested_ffn_contract(const bf16_t* __restrict__ h_in,
                         const int*    __restrict__ token_mask,
                         const float*  __restrict__ W2,
                         const float*  __restrict__ b2,
                         float*        __restrict__ out,
                         int n_tok)
{
    __shared__ __align__(16) bf16_t As[TM][LDK];
    __shared__ __align__(16) bf16_t Bs[TN][LDK];
    __shared__ float s_bias[TN];
    __shared__ int   s_minshift;

    const int tid      = threadIdx.x;
    const int tok_base = blockIdx.y * TM;
    const int d_base   = blockIdx.x * TN;

    if (tid == 0) s_minshift = 3;
    __syncthreads();
    if (tid < TM) atomicMin(&s_minshift, 3 - token_mask[tok_base + tid]);
    if (tid < TN) s_bias[tid] = b2[d_base + tid];
    __syncthreads();

    const int kmax = H_DIM >> s_minshift;  // h rows are zero beyond max d_hid

    const int lane = tid & 31;
    const int half = lane >> 4;
    const int l16  = lane & 15;
    const int wave = tid >> 5;
    const int wm   = wave & 3;
    const int wn   = wave >> 2;

    v8f acc[2][4] = {};

    for (int k0 = 0; k0 < kmax; k0 += TK) {
#if __has_builtin(__builtin_amdgcn_tensor_load_to_lds)
        // ---- TDM: DMA the 128x64 bf16 h tile into the padded LDS layout ----
        if (wave == 0) {
            const unsigned long long ga =
                (unsigned long long)(const void*)(h_in + (size_t)tok_base * H_DIM + k0);
            const unsigned lds_off = (unsigned)(size_t)(void*)&As[0][0];
            u32x4 g0;
            g0[0] = 1u;                                  // count=1, user mode
            g0[1] = lds_off;                             // LDS byte address
            g0[2] = (unsigned)(ga & 0xFFFFFFFFu);        // global_addr[31:0]
            g0[3] = (unsigned)((ga >> 32) & 0x01FFFFFFu) // global_addr[56:32]
                  | (2u << 30);                          // type = 2 (image)
            i32x8 g1;
            g1[0] = (1 << 16)       // data_size = 2 bytes (bf16)
                  | (1 << 20)       // pad_enable (reproduce LDK padding)
                  | (4 << 22)       // pad_interval: 32 DWORDs (= 64 bf16 row)
                  | (3 << 25);      // pad_amount: 4 DWORDs (= 8 bf16 pad)
            g1[1] = (int)((H_DIM & 0xFFFF) << 16);                       // tensor_dim0 lo
            g1[2] = (int)(((H_DIM >> 16) & 0xFFFF) |
                          ((unsigned)(n_tok & 0xFFFF) << 16));           // dim0 hi | dim1 lo
            g1[3] = (int)((((unsigned)n_tok >> 16) & 0xFFFF) | (TK << 16)); // dim1 hi | tile_dim0=64
            g1[4] = TM;                                                  // tile_dim1=128 (dim2=0)
            g1[5] = H_DIM;                                               // tensor_dim0_stride lo
            g1[6] = 0;
            g1[7] = 0;
            const i32x4 gz4 = {0, 0, 0, 0};
            const i32x8 gz8 = {0, 0, 0, 0, 0, 0, 0, 0};
            // 6-arg toolchain variant: (g0, g1, g2, g3, g4, cpol)
            __builtin_amdgcn_tensor_load_to_lds(g0, g1, gz4, gz4, gz8, 0);
            __builtin_amdgcn_s_wait_tensorcnt(0);        // land before barrier
        }
#else
        // Fallback: vector copy of the bf16 h tile (128-bit chunks).
#pragma unroll
        for (int i = 0; i < 4; ++i) {
            const int idx = tid + i * THREADS;           // 1024 x uint4
            const int row = idx >> 3;
            const int c8  = (idx & 7) << 3;
            *(uint4*)&As[row][c8] =
                *(const uint4*)(h_in + (size_t)(tok_base + row) * H_DIM + k0 + c8);
        }
#endif
        // B: W2 tile, fp32 -> bf16.
#pragma unroll
        for (int i = 0; i < 8; ++i) {
            const int idx = tid + i * THREADS;           // 2048 float4 per tile
            const int row = idx >> 4;
            const int c4  = (idx & 15) << 2;
            const float4 vb =
                *(const float4*)(W2 + (size_t)(d_base + row) * H_DIM + k0 + c4);
            Bs[row][c4 + 0] = (bf16_t)vb.x;
            Bs[row][c4 + 1] = (bf16_t)vb.y;
            Bs[row][c4 + 2] = (bf16_t)vb.z;
            Bs[row][c4 + 3] = (bf16_t)vb.w;
        }
        if (k0 + TK < kmax) {
            const int prow = tid >> 1, pcol = (tid & 1) << 5;
            __builtin_prefetch(W2 + (size_t)(d_base + prow) * H_DIM + k0 + TK + pcol, 0, 1);
        }
        __syncthreads();
        wmma_tile_step(As, Bs, wm, wn, half, l16, acc);
        __syncthreads();
    }

    // Epilogue: full output plus bias (fp32).
#pragma unroll
    for (int mi = 0; mi < 2; ++mi) {
        const int lm = wm * 32 + mi * 16 + half * 8;
#pragma unroll
        for (int r = 0; r < 8; ++r) {
            float* orow = out + (size_t)(tok_base + lm + r) * D_DIM + d_base;
#pragma unroll
            for (int ni = 0; ni < 4; ++ni) {
                const int ln = wn * 64 + ni * 16 + l16;
                orow[ln] = acc[mi][ni][r] + s_bias[ln];
            }
        }
    }
}

// ---------------------------------------------------------------------------
extern "C" void kernel_launch(void* const* d_in, const int* in_sizes, int n_in,
                              void* d_out, int out_size, void* d_ws, size_t ws_size,
                              hipStream_t stream) {
    const float* x          = (const float*)d_in[0];
    const int*   token_mask = (const int*)  d_in[1];
    const float* W1         = (const float*)d_in[2];
    const float* b1         = (const float*)d_in[3];
    const float* W2         = (const float*)d_in[4];
    const float* b2         = (const float*)d_in[5];
    float*       out        = (float*)d_out;
    bf16_t*      h_ws       = (bf16_t*)d_ws;   // N*H bf16 = 128 MB scratch

    const int n_tok = in_sizes[1];             // B*T = 16384 (multiple of TM)

    dim3 g1(H_DIM / TN, n_tok / TM);
    dim3 g2(D_DIM / TN, n_tok / TM);
    nested_ffn_expand  <<<g1, THREADS, 0, stream>>>(x, token_mask, W1, b1, h_ws);
    nested_ffn_contract<<<g2, THREADS, 0, stream>>>(h_ws, token_mask, W2, b2, out, n_tok);
}
